// SparseAttention_69947837383377
// MI455X (gfx1250) — matrix operand order repaired and checked by reading, not compiled, not run on hardware
//
#include <hip/hip_runtime.h>

#define D_MODEL 1024
#define NHEADS  16
#define HDIM    64
#define WIN     64
#define BATCH   2
#define SEQL    2048
#define MTOK    (BATCH * SEQL)   // 4096 token rows
#define NKEY    160              // padded key window (10 tiles of 16, 5 k-steps of 32)

#define LOG2E 1.4426950408889634f

typedef __bf16 bf16;
typedef __attribute__((ext_vector_type(16))) __bf16 v16bf;
typedef __attribute__((ext_vector_type(8)))  __bf16 v8bf;
typedef __attribute__((ext_vector_type(8)))  float  v8f;

static __device__ __forceinline__ v16bf load_a_frag(const bf16* p) {
    // A 16x32 frag in this file's contiguous k order (B same): lane holds K = h*16+0..15, caller
    // passes p already offset by (row, k0 + half*16). Two 16B chunks, contiguous.
    v8bf lo = *(const v8bf*)p;
    v8bf hi = *(const v8bf*)(p + 8);
    return __builtin_shufflevector(lo, hi, 0, 1, 2, 3, 4, 5, 6, 7,
                                            8, 9, 10, 11, 12, 13, 14, 15);
}

static __device__ __forceinline__ v8f wmma_bf16(v16bf a, v16bf b, v8f c) {
    return __builtin_amdgcn_wmma_f32_16x16x32_bf16(
        /*neg_a=*/false, a, /*neg_b=*/false, b,
        /*c_mod=*/(short)0, c, /*reuse_a=*/false, /*reuse_b=*/false);
}

static __device__ __forceinline__ int iclamp(int v, int lo, int hi) {
    return v < lo ? lo : (v > hi ? hi : v);
}

// ---------------------------------------------------------------------------
// fp32 -> bf16 elementwise convert
// ---------------------------------------------------------------------------
__global__ void cvt_f32_bf16(const float* __restrict__ in, bf16* __restrict__ out, int n) {
    int i = blockIdx.x * blockDim.x + threadIdx.x;
    if (i < n) out[i] = (bf16)in[i];
}

// ---------------------------------------------------------------------------
// W [K,N] fp32 -> Wt [N,K] bf16 (LDS tiled transpose)
// ---------------------------------------------------------------------------
__global__ void transpose_cvt(const float* __restrict__ W, bf16* __restrict__ Wt,
                              int K, int N) {
    __shared__ float tile[32][33];
    int n0 = blockIdx.x * 32, k0 = blockIdx.y * 32;
    int tx = threadIdx.x & 31, ty = threadIdx.x >> 5;   // 256 thr = 32x8
    for (int r = ty; r < 32; r += 8)
        tile[r][tx] = W[(size_t)(k0 + r) * N + n0 + tx];
    __syncthreads();
    for (int r = ty; r < 32; r += 8)
        Wt[(size_t)(n0 + r) * K + k0 + tx] = (bf16)tile[tx][r];
}

// ---------------------------------------------------------------------------
// GEMM: C[m,n] = sum_k A[m,k] * Bt[n,k], A/Bt bf16, acc f32 via WMMA.
// M=4096, N=1024, K=1024 hardcoded. Block 256 thr (8 waves), block tile
// 128x256, wave tile 64x64 (4x4 WMMA accumulators -> 16 WMMA / 8 frag loads
// per K-step for ~2x the arithmetic intensity of a 32x64 tile).
// MODE 0: fp32 C[M,N]
// MODE 1: bf16 [b,h,s,d]   (Q, K storage)
// MODE 2: bf16 [b,h,d,s]   (V transposed storage)
// ---------------------------------------------------------------------------
template <int MODE>
__global__ void __launch_bounds__(256)
gemm_bf16(const bf16* __restrict__ A, const bf16* __restrict__ Bt, void* __restrict__ C) {
    const int K = 1024;
    int lane = threadIdx.x & 31;
    int wave = threadIdx.x >> 5;
    int half = lane >> 4, l16 = lane & 15;
    int wm = wave & 1, wn = wave >> 1;     // 2 waves along M, 4 along N
    int m0 = blockIdx.x * 128 + wm * 64;   // gridDim.x = 32
    int n0 = blockIdx.y * 256 + wn * 64;   // gridDim.y = 4

    v8f acc[4][4];
#pragma unroll
    for (int i = 0; i < 4; ++i)
#pragma unroll
        for (int j = 0; j < 4; ++j) acc[i][j] = (v8f){};

    const bf16* Ab = A + (size_t)(m0 + l16) * K + half * 16;
    const bf16* Bb = Bt + (size_t)(n0 + l16) * K + half * 16;

    for (int k0 = 0; k0 < K; k0 += 32) {
        v16bf af[4], bfr[4];
#pragma unroll
        for (int mt = 0; mt < 4; ++mt)
            af[mt] = load_a_frag(Ab + (size_t)(mt * 16) * K + k0);
#pragma unroll
        for (int nt = 0; nt < 4; ++nt)
            bfr[nt] = *(const v16bf*)(Bb + (size_t)(nt * 16) * K + k0);
#pragma unroll
        for (int mt = 0; mt < 4; ++mt)
#pragma unroll
            for (int nt = 0; nt < 4; ++nt)
                acc[mt][nt] = wmma_bf16(af[mt], bfr[nt], acc[mt][nt]);
    }

#pragma unroll
    for (int mt = 0; mt < 4; ++mt)
#pragma unroll
        for (int nt = 0; nt < 4; ++nt)
#pragma unroll
            for (int r = 0; r < 8; ++r) {
                int m = m0 + mt * 16 + r + (half << 3);
                int n = n0 + nt * 16 + l16;
                float v = acc[mt][nt][r];
                if (MODE == 0) {
                    ((float*)C)[(size_t)m * D_MODEL + n] = v;
                } else {
                    int b = m >> 11, s = m & (SEQL - 1);
                    int h = n >> 6, d = n & (HDIM - 1);
                    if (MODE == 1)
                        ((bf16*)C)[(((size_t)(b * NHEADS + h) * SEQL) + s) * HDIM + d] = (bf16)v;
                    else
                        ((bf16*)C)[(((size_t)(b * NHEADS + h) * HDIM) + d) * SEQL + s] = (bf16)v;
                }
            }
}

// ---------------------------------------------------------------------------
// Banded attention. Grid = B*H*(SEQL/64) blocks of 128 threads (4 waves).
// Each wave owns a 16-query tile; window keys [q0-64, q0+79] padded to 160.
// Scores are pre-scaled by 0.125*log2(e) so softmax uses native exp2.
// ---------------------------------------------------------------------------
__global__ void __launch_bounds__(128)
attn_band(const bf16* __restrict__ Qb, const bf16* __restrict__ Kb,
          const bf16* __restrict__ Vt, bf16* __restrict__ O) {
    __shared__ __align__(16) float s_sc[4][16][NKEY];
    __shared__ __align__(16) bf16  s_p[4][16][NKEY];
    __shared__ float s_rinv[4][16];

    int w = threadIdx.x >> 5;
    int lane = threadIdx.x & 31;
    int half = lane >> 4, l16 = lane & 15;
    int blk = blockIdx.x;             // 0..1023
    int bh = blk >> 5;                // b*16 + h
    int q0 = ((blk & 31) << 6) + w * 16;

    const bf16* Qbase = Qb + (size_t)bh * SEQL * HDIM;
    const bf16* Kbase = Kb + (size_t)bh * SEQL * HDIM;
    const bf16* Vbase = Vt + (size_t)bh * HDIM * SEQL;

    // Q fragments (A-matrix, 16 queries x 32 hd-channels, two hd halves)
    int qrow = q0 + l16;
    v16bf qf0 = load_a_frag(Qbase + (size_t)qrow * HDIM + half * 16);
    v16bf qf1 = load_a_frag(Qbase + (size_t)qrow * HDIM + 32 + half * 16);

    int kstart = q0 - WIN;
    const float scale = 0.125f * LOG2E;   // 1/sqrt(64) folded with log2(e)

    // ---- scores: S = (Q K^T) * scale (log2 domain), band-masked, to LDS ----
    for (int kt = 0; kt < 10; ++kt) {
        int j0 = kstart + kt * 16;
        int jl = iclamp(j0 + l16, 0, SEQL - 1);
        v16bf kf0 = *(const v16bf*)(Kbase + (size_t)jl * HDIM + half * 16);
        v16bf kf1 = *(const v16bf*)(Kbase + (size_t)jl * HDIM + 32 + half * 16);
        v8f c = (v8f){};
        c = wmma_bf16(qf0, kf0, c);
        c = wmma_bf16(qf1, kf1, c);
#pragma unroll
        for (int r = 0; r < 8; ++r) {
            int m = r + (half << 3);
            int q = q0 + m;
            int j = j0 + l16;
            float v = c[r] * scale;
            if (j < 0 || j >= SEQL || (j - q) > WIN || (q - j) > WIN) v = -1e30f;
            s_sc[w][m][kt * 16 + l16] = v;
        }
    }
    __syncthreads();

    // ---- row softmax in log2 domain (unnormalized P; 1/rowsum kept) ----
    {
        int row = l16;
        const float* src = &s_sc[w][row][half * 80];
        float mx = -1e30f;
        for (int t = 0; t < 80; ++t) mx = fmaxf(mx, src[t]);
        mx = fmaxf(mx, __shfl_xor(mx, 16));
        float sum = 0.f;
        bf16* dst = &s_p[w][row][half * 80];
        for (int t = 0; t < 80; ++t) {
            float e = exp2f(src[t] - mx);   // native v_exp_f32
            sum += e;
            dst[t] = (bf16)e;
        }
        sum += __shfl_xor(sum, 16);
        if (half == 0) s_rinv[w][row] = 1.0f / sum;
    }
    __syncthreads();

    // ---- O = P @ V ----
    v8f o[4] = {(v8f){}, (v8f){}, (v8f){}, (v8f){}};
    for (int step = 0; step < 5; ++step) {
        int k0 = step * 32;
        v16bf pf = load_a_frag(&s_p[w][l16][k0 + half * 16]);
        int j0h = iclamp(kstart + k0 + half * 16, 0, SEQL - 16);
#pragma unroll
        for (int nt = 0; nt < 4; ++nt) {
            int d = nt * 16 + l16;
            v16bf vf = *(const v16bf*)(Vbase + (size_t)d * SEQL + j0h);
            o[nt] = wmma_bf16(pf, vf, o[nt]);
        }
    }

#pragma unroll
    for (int nt = 0; nt < 4; ++nt)
#pragma unroll
        for (int r = 0; r < 8; ++r) {
            int m = r + (half << 3);
            float v = o[nt][r] * s_rinv[w][m];
            int tok = (bh >> 4) * SEQL + q0 + m;
            int col = (bh & 15) * HDIM + nt * 16 + l16;
            O[(size_t)tok * D_MODEL + col] = (bf16)v;
        }
}

// ---------------------------------------------------------------------------
// Host launcher
// ---------------------------------------------------------------------------
extern "C" void kernel_launch(void* const* d_in, const int* in_sizes, int n_in,
                              void* d_out, int out_size, void* d_ws, size_t ws_size,
                              hipStream_t stream) {
    const float* x  = (const float*)d_in[0];
    const float* Wq = (const float*)d_in[1];
    const float* Wk = (const float*)d_in[3];
    const float* Wv = (const float*)d_in[5];
    const float* Wo = (const float*)d_in[7];
    float* out = (float*)d_out;

    char* ws = (char*)d_ws;
    const size_t MB = 1024ull * 1024ull;
    bf16* xb    = (bf16*)(ws + 0);         // 8 MB  (4096x1024)
    bf16* wqT   = (bf16*)(ws + 8 * MB);    // 2 MB
    bf16* wkT   = (bf16*)(ws + 10 * MB);   // 2 MB
    bf16* wvT   = (bf16*)(ws + 12 * MB);   // 2 MB
    bf16* woT   = (bf16*)(ws + 14 * MB);   // 2 MB
    bf16* Qb    = (bf16*)(ws + 16 * MB);   // 8 MB  [b,h,s,d]
    bf16* Kb    = (bf16*)(ws + 24 * MB);   // 8 MB  [b,h,s,d]
    bf16* Vt    = (bf16*)(ws + 32 * MB);   // 8 MB  [b,h,d,s]
    bf16* attnb = (bf16*)(ws + 40 * MB);   // 8 MB  (4096x1024)

    // 1) convert x to bf16
    {
        int n = MTOK * D_MODEL;
        cvt_f32_bf16<<<(n + 255) / 256, 256, 0, stream>>>(x, xb, n);
    }
    // 2) transpose-convert weights
    {
        dim3 g(D_MODEL / 32, D_MODEL / 32);
        transpose_cvt<<<g, 256, 0, stream>>>(Wq, wqT, D_MODEL, D_MODEL);
        transpose_cvt<<<g, 256, 0, stream>>>(Wk, wkT, D_MODEL, D_MODEL);
        transpose_cvt<<<g, 256, 0, stream>>>(Wv, wvT, D_MODEL, D_MODEL);
        transpose_cvt<<<g, 256, 0, stream>>>(Wo, woT, D_MODEL, D_MODEL);
    }
    // 3) QKV projections (WMMA GEMMs)
    {
        dim3 g(MTOK / 128, D_MODEL / 256);
        gemm_bf16<1><<<g, 256, 0, stream>>>(xb, wqT, (void*)Qb);
        gemm_bf16<1><<<g, 256, 0, stream>>>(xb, wkT, (void*)Kb);
        gemm_bf16<2><<<g, 256, 0, stream>>>(xb, wvT, (void*)Vt);
    }
    // 4) banded attention
    attn_band<<<BATCH * NHEADS * (SEQL / 64), 128, 0, stream>>>(Qb, Kb, Vt, attnb);
    // 5) output projection -> fp32 d_out
    {
        dim3 g(MTOK / 128, D_MODEL / 256);
        gemm_bf16<0><<<g, 256, 0, stream>>>(attnb, woT, (void*)out);
    }
}
